// BiLSTM_CRF_85152021610986
// MI455X (gfx1250) — matrix operand order repaired and checked by reading, not compiled, not run on hardware
//
#include <hip/hip_runtime.h>

// ---------------------------------------------------------------------------
// BiLSTM-CRF NLL on MI455X (gfx1250, wave32, WMMA bf16 -> f32 accumulate)
//
//   1. Convert weights to bf16; gather embeddings to bf16 (L,B,E).
//   2. One big WMMA GEMM: xproj[dir] = x @ W_ih[dir]^T + b[dir]  (f32, all t).
//      2x2 register-blocked: 32x32 output per wave, 4 WMMA per K-step.
//   3. 512 sequential steps; each launch handles BOTH directions (fwd t=s,
//      bwd t=L-1-s): gates = xproj[t] + h_prev @ W_hh^T (WMMA, 2x2 blocked),
//      then fused elementwise LSTM cell update (f32 state, bf16 h out).
//   4. feats = [hf|hb] @ W_out^T + b_out via WMMA (N padded 10 -> 16,
//      2x1 M-blocked so the B fragment is reused).
//   5. CRF gold score + forward algorithm, one thread per batch item.
// ---------------------------------------------------------------------------

constexpr int Bc  = 64;     // batch
constexpr int Lc  = 512;    // sequence length
constexpr int Ec  = 512;    // embedding dim
constexpr int Hc  = 512;    // hidden per direction
constexpr int G4  = 2048;   // 4*H gate width
constexpr int HDc = 1024;   // 2*H
constexpr int Tc  = 10;     // tags
constexpr size_t LB = (size_t)Lc * Bc;   // 32768 rows

typedef __attribute__((ext_vector_type(16))) __bf16 v16bf;
typedef __attribute__((ext_vector_type(8)))  float  v8f;

// ---------------------------------------------------------------------------
// WMMA helpers (ISA 7.12.2 fragment layouts, wave32)
// ---------------------------------------------------------------------------

__device__ __forceinline__ v8f wmma_bf16(v16bf a, v16bf b, v8f c) {
  // 8 args: (neg_a, A, neg_b, B, c_mod, C, reuse_a, reuse_b)
  return __builtin_amdgcn_wmma_f32_16x16x32_bf16(false, a, false, b,
                                                 (short)0, c, false, false);
}

// A fragment: 16x32 (MxK), row-major source with leading dim lda (elements).
// Lane L (0-15): row M=L, K = {kbase..kbase+7, kbase+16..kbase+23}
// Lane L+16   : row M=L, K = {kbase+8..kbase+15, kbase+24..kbase+31}
__device__ __forceinline__ v16bf load_a_tile(const __bf16* __restrict__ A,
                                             int lda, int kbase) {
  const int lane = threadIdx.x & 31;
  const int row  = lane & 15;
  const int kh   = (lane >> 4) << 3;  // 0 or 8
  const __bf16* p = A + (size_t)row * lda + kbase + kh;
  union { v16bf v; __bf16 e[16]; } a;
#pragma unroll
  for (int i = 0; i < 8; ++i) a.e[i] = p[i];
#pragma unroll
  for (int i = 0; i < 8; ++i) a.e[8 + i] = p[16 + i];
  return a.v;
}

// B fragment: 32x16 (KxN) where B[k][n] = W[n][k], W row-major with leading
// dim ldw. Lane L (0-15): col N=L, K = {kbase..kbase+7, kbase+16..kbase+23} ;
// lane L+16: K = {kbase+8..kbase+15, kbase+24..kbase+31}.
__device__ __forceinline__ v16bf load_bt_tile(const __bf16* __restrict__ W,
                                              int ldw, int nbase, int kbase) {
  const int lane = threadIdx.x & 31;
  const int n  = nbase + (lane & 15);
  const int k0 = kbase + ((lane >> 4) << 3);  // 0 or 8
  const __bf16* p = W + (size_t)n * ldw + k0;
  union { v16bf v; __bf16 e[16]; } b;
#pragma unroll
  for (int i = 0; i < 8; ++i) b.e[i] = p[i];
#pragma unroll
  for (int i = 0; i < 8; ++i) b.e[8 + i] = p[16 + i];
  return b.v;
}

__device__ __forceinline__ float sigmoidf(float x) {
  return 1.f / (1.f + __expf(-x));
}

// ---------------------------------------------------------------------------
// Prep kernels
// ---------------------------------------------------------------------------

__global__ void f32_to_bf16_kernel(const float* __restrict__ s,
                                   __bf16* __restrict__ d, int n) {
  for (int i = blockIdx.x * blockDim.x + threadIdx.x; i < n;
       i += gridDim.x * blockDim.x)
    d[i] = (__bf16)s[i];
}

// W_out (T=10, HD=1024) -> bf16 padded to 16 rows (rows 10..15 zero)
__global__ void wout_pad_kernel(const float* __restrict__ w,
                                __bf16* __restrict__ d) {
  const int n = 16 * HDc;
  for (int i = blockIdx.x * blockDim.x + threadIdx.x; i < n;
       i += gridDim.x * blockDim.x) {
    const int row = i >> 10, k = i & (HDc - 1);
    d[i] = (row < Tc) ? (__bf16)w[row * HDc + k] : (__bf16)0.f;
  }
}

__global__ void init_state_kernel(__bf16* hF, __bf16* hB, float* cF, float* cB) {
  const int n = Bc * Hc;
  for (int i = blockIdx.x * blockDim.x + threadIdx.x; i < n;
       i += gridDim.x * blockDim.x) {
    hF[i] = (__bf16)0.f; hB[i] = (__bf16)0.f;
    cF[i] = 0.f;         cB[i] = 0.f;
  }
}

// x[(l*B+b)*E + e] = emb[sentence[b*L+l]][e]  (bf16)
__global__ void embed_kernel(const int* __restrict__ sent,
                             const float* __restrict__ emb,
                             __bf16* __restrict__ x) {
  const size_t total = LB * Ec;
  for (size_t idx = blockIdx.x * (size_t)blockDim.x + threadIdx.x; idx < total;
       idx += (size_t)gridDim.x * blockDim.x) {
    const int e  = (int)(idx & (Ec - 1));
    const int rb = (int)(idx >> 9);       // row = l*64 + b
    const int b  = rb & (Bc - 1);
    const int l  = rb >> 6;
    const int tok = sent[b * Lc + l];
    x[idx] = (__bf16)emb[(size_t)tok * Ec + e];
  }
}

// ---------------------------------------------------------------------------
// xproj[dir] = x @ W_ih[dir]^T + bias   (LB x 2048, K = 512), WMMA
// 2x2 register blocking: each wave computes a 32x32 patch.
// grid (8, LB/32, 2), block 256 (8 waves)
// ---------------------------------------------------------------------------
__global__ void xproj_kernel(const __bf16* __restrict__ x,
                             const __bf16* __restrict__ wihF,
                             const __bf16* __restrict__ wihB,
                             const float* __restrict__ bF,
                             const float* __restrict__ bB,
                             float* __restrict__ xprojF,
                             float* __restrict__ xprojB) {
  const int dir = blockIdx.z;
  const __bf16* w   = dir ? wihB : wihF;
  const float* bias = dir ? bB : bF;
  float* out        = dir ? xprojB : xprojF;

  const int wave = threadIdx.x >> 5;
  const int n2   = blockIdx.x * 8 + wave;   // 0..63  (32-col groups)
  const int m2   = blockIdx.y;              // 0..1023 (32-row groups)
  const int lane = threadIdx.x & 31;
  const int col0 = n2 * 32 + (lane & 15);
  const int r0   = m2 * 32 + ((lane >> 4) << 3);

  union { v8f v; float e[8]; } acc00, acc01, acc10, acc11;
  const float bv0 = bias[col0], bv1 = bias[col0 + 16];
#pragma unroll
  for (int r = 0; r < 8; ++r) {
    acc00.e[r] = bv0; acc01.e[r] = bv1;
    acc10.e[r] = bv0; acc11.e[r] = bv1;
  }

  const __bf16* arow = x + (size_t)m2 * 32 * Ec;
  for (int kt = 0; kt < Ec / 32; ++kt) {
    const int kb = kt * 32;
    v16bf a0 = load_a_tile(arow, Ec, kb);
    v16bf a1 = load_a_tile(arow + (size_t)16 * Ec, Ec, kb);
    v16bf b0 = load_bt_tile(w, Ec, n2 * 32, kb);
    v16bf b1 = load_bt_tile(w, Ec, n2 * 32 + 16, kb);
    acc00.v = wmma_bf16(a0, b0, acc00.v);
    acc01.v = wmma_bf16(a0, b1, acc01.v);
    acc10.v = wmma_bf16(a1, b0, acc10.v);
    acc11.v = wmma_bf16(a1, b1, acc11.v);
  }
#pragma unroll
  for (int r = 0; r < 8; ++r) {
    out[(size_t)(r0 + r) * G4 + col0]           = acc00.e[r];
    out[(size_t)(r0 + r) * G4 + col0 + 16]      = acc01.e[r];
    out[(size_t)(r0 + 16 + r) * G4 + col0]      = acc10.e[r];
    out[(size_t)(r0 + 16 + r) * G4 + col0 + 16] = acc11.e[r];
  }
}

// ---------------------------------------------------------------------------
// Per-step gates: gates[dir] = xproj[dir][t] + h_prev[dir] @ W_hh[dir]^T
// 2x2 register blocking: 32x32 per wave. grid (64, 2, 2), block 32.
// fwd: t = s ; bwd: t = L-1-s
// ---------------------------------------------------------------------------
__global__ void lstm_gates_kernel(const float* __restrict__ xprojF,
                                  const float* __restrict__ xprojB,
                                  const __bf16* __restrict__ hFst,
                                  const __bf16* __restrict__ hBst,
                                  const __bf16* __restrict__ whhF,
                                  const __bf16* __restrict__ whhB,
                                  float* __restrict__ gatesF,
                                  float* __restrict__ gatesB, int s) {
  const int dir = blockIdx.z;
  const int tt  = dir ? (Lc - 1 - s) : s;
  const float*  xp = (dir ? xprojB : xprojF) + (size_t)tt * Bc * G4;
  const __bf16* h  = dir ? hBst : hFst;
  const __bf16* w  = dir ? whhB : whhF;
  float* gates     = dir ? gatesB : gatesF;

  const int n2   = blockIdx.x;              // 0..63 (32-col groups)
  const int m2   = blockIdx.y;              // 0..1  (32-row groups)
  const int lane = threadIdx.x & 31;
  const int col0 = n2 * 32 + (lane & 15);
  const int r0   = m2 * 32 + ((lane >> 4) << 3);

  union { v8f v; float e[8]; } acc00, acc01, acc10, acc11;
#pragma unroll
  for (int r = 0; r < 8; ++r) {
    acc00.e[r] = xp[(size_t)(r0 + r) * G4 + col0];
    acc01.e[r] = xp[(size_t)(r0 + r) * G4 + col0 + 16];
    acc10.e[r] = xp[(size_t)(r0 + 16 + r) * G4 + col0];
    acc11.e[r] = xp[(size_t)(r0 + 16 + r) * G4 + col0 + 16];
  }

  const __bf16* arow = h + (size_t)m2 * 32 * Hc;
  for (int kt = 0; kt < Hc / 32; ++kt) {
    const int kb = kt * 32;
    v16bf a0 = load_a_tile(arow, Hc, kb);
    v16bf a1 = load_a_tile(arow + (size_t)16 * Hc, Hc, kb);
    v16bf b0 = load_bt_tile(w, Hc, n2 * 32, kb);
    v16bf b1 = load_bt_tile(w, Hc, n2 * 32 + 16, kb);
    acc00.v = wmma_bf16(a0, b0, acc00.v);
    acc01.v = wmma_bf16(a0, b1, acc01.v);
    acc10.v = wmma_bf16(a1, b0, acc10.v);
    acc11.v = wmma_bf16(a1, b1, acc11.v);
  }
#pragma unroll
  for (int r = 0; r < 8; ++r) {
    gates[(size_t)(r0 + r) * G4 + col0]           = acc00.e[r];
    gates[(size_t)(r0 + r) * G4 + col0 + 16]      = acc01.e[r];
    gates[(size_t)(r0 + 16 + r) * G4 + col0]      = acc10.e[r];
    gates[(size_t)(r0 + 16 + r) * G4 + col0 + 16] = acc11.e[r];
  }
}

// ---------------------------------------------------------------------------
// Elementwise LSTM cell update (both directions), grid (128, 2) x 256
// ---------------------------------------------------------------------------
__global__ void lstm_cell_kernel(const float* __restrict__ gF,
                                 const float* __restrict__ gB,
                                 float* __restrict__ cF, float* __restrict__ cB,
                                 __bf16* __restrict__ hFst, __bf16* __restrict__ hBst,
                                 __bf16* __restrict__ hfall, __bf16* __restrict__ hball,
                                 int s) {
  const int dir = blockIdx.y;
  const int idx = blockIdx.x * blockDim.x + threadIdx.x;  // < 64*512
  const int b = idx >> 9, j = idx & (Hc - 1);
  const float* g  = dir ? gB : gF;
  float* c        = dir ? cB : cF;
  __bf16* hst     = dir ? hBst : hFst;
  __bf16* hall    = dir ? hball : hfall;

  const size_t base = (size_t)b * G4;
  const float iv = sigmoidf(g[base + j]);
  const float fv = sigmoidf(g[base + Hc + j]);
  const float gv = tanhf(g[base + 2 * Hc + j]);
  const float ov = sigmoidf(g[base + 3 * Hc + j]);
  const float cn = fv * c[idx] + iv * gv;
  const float hn = ov * tanhf(cn);
  c[idx]  = cn;
  hst[idx] = (__bf16)hn;
  const int tt = dir ? (Lc - 1 - s) : s;
  hall[((size_t)tt * Bc + b) * Hc + j] = (__bf16)hn;
}

// ---------------------------------------------------------------------------
// feats = [hf|hb] @ W_out^T + b_out  (LB x 10, K = 1024), WMMA (N padded 16)
// 2x1 M-blocking (B fragment reused). grid (LB/32) x 32
// ---------------------------------------------------------------------------
__global__ void feats_kernel(const __bf16* __restrict__ hfall,
                             const __bf16* __restrict__ hball,
                             const __bf16* __restrict__ wout,  // [16][1024]
                             const float* __restrict__ bout,
                             float* __restrict__ feats) {
  const int m2   = blockIdx.x;              // 0..1023 (32-row groups)
  const int lane = threadIdx.x & 31;
  const int n    = lane & 15;
  const int r0   = m2 * 32 + ((lane >> 4) << 3);

  union { v8f v; float e[8]; } acc0, acc1;
  const float bv = (n < Tc) ? bout[n] : 0.f;
#pragma unroll
  for (int r = 0; r < 8; ++r) { acc0.e[r] = bv; acc1.e[r] = bv; }

  const __bf16* af = hfall + (size_t)m2 * 32 * Hc;
  const __bf16* ab = hball + (size_t)m2 * 32 * Hc;
  for (int kt = 0; kt < Hc / 32; ++kt) {
    const int kb = kt * 32;
    v16bf b = load_bt_tile(wout, HDc, 0, kb);
    v16bf a0 = load_a_tile(af, Hc, kb);
    v16bf a1 = load_a_tile(af + (size_t)16 * Hc, Hc, kb);
    acc0.v = wmma_bf16(a0, b, acc0.v);
    acc1.v = wmma_bf16(a1, b, acc1.v);
  }
  for (int kt = 0; kt < Hc / 32; ++kt) {
    const int kb = kt * 32;
    v16bf b = load_bt_tile(wout, HDc, 0, Hc + kb);
    v16bf a0 = load_a_tile(ab, Hc, kb);
    v16bf a1 = load_a_tile(ab + (size_t)16 * Hc, Hc, kb);
    acc0.v = wmma_bf16(a0, b, acc0.v);
    acc1.v = wmma_bf16(a1, b, acc1.v);
  }
  if (n < Tc) {
#pragma unroll
    for (int r = 0; r < 8; ++r) {
      feats[(size_t)(r0 + r) * Tc + n]      = acc0.e[r];
      feats[(size_t)(r0 + 16 + r) * Tc + n] = acc1.e[r];
    }
  }
}

// ---------------------------------------------------------------------------
// CRF NLL: one thread per batch item; gold score + forward algorithm (T=10)
// ---------------------------------------------------------------------------
__global__ void crf_kernel(const float* __restrict__ feats,
                           const int* __restrict__ tags,
                           const unsigned char* __restrict__ mask,
                           const float* __restrict__ start,
                           const float* __restrict__ endt,
                           const float* __restrict__ trans,
                           float* __restrict__ out) {
  __shared__ float str[Tc * Tc];
  __shared__ float red_llh[Bc];
  __shared__ float red_msk[Bc];
  for (int i = threadIdx.x; i < Tc * Tc; i += blockDim.x) str[i] = trans[i];
  __syncthreads();

  const int b = threadIdx.x;
  float alpha[Tc];
  int tprev = tags[b * Lc];
  float score = start[tprev] + feats[b * Tc + tprev];
  float msum = mask[b * Lc] ? 1.f : 0.f;
#pragma unroll
  for (int j = 0; j < Tc; ++j) alpha[j] = start[j] + feats[b * Tc + j];

  for (int t = 1; t < Lc; ++t) {
    const float* em = feats + ((size_t)t * Bc + b) * Tc;
    const int tcur = tags[b * Lc + t];
    const float m = mask[b * Lc + t] ? 1.f : 0.f;
    score += m * (str[tprev * Tc + tcur] + em[tcur]);
    tprev = tcur;

    float nxt[Tc];
#pragma unroll
    for (int j = 0; j < Tc; ++j) {
      float mx = -1e30f;
#pragma unroll
      for (int i = 0; i < Tc; ++i)
        mx = fmaxf(mx, alpha[i] + str[i * Tc + j]);
      float ssum = 0.f;
#pragma unroll
      for (int i = 0; i < Tc; ++i)
        ssum += __expf(alpha[i] + str[i * Tc + j] - mx);
      nxt[j] = mx + __logf(ssum) + em[j];
    }
    if (m > 0.f) {
#pragma unroll
      for (int j = 0; j < Tc; ++j) alpha[j] = nxt[j];
    }
    msum += m;
  }
  const int last = (int)msum - 1;
  score += endt[tags[b * Lc + last]];

  float mx = -1e30f;
#pragma unroll
  for (int j = 0; j < Tc; ++j) mx = fmaxf(mx, alpha[j] + endt[j]);
  float ssum = 0.f;
#pragma unroll
  for (int j = 0; j < Tc; ++j) ssum += __expf(alpha[j] + endt[j] - mx);
  const float denom = mx + __logf(ssum);

  red_llh[b] = score - denom;
  red_msk[b] = msum;
  __syncthreads();
  if (b == 0) {
    float sl = 0.f, sm = 0.f;
    for (int i = 0; i < Bc; ++i) { sl += red_llh[i]; sm += red_msk[i]; }
    out[0] = -(sl / sm);
  }
}

// ---------------------------------------------------------------------------
// Host launcher
// ---------------------------------------------------------------------------
extern "C" void kernel_launch(void* const* d_in, const int* in_sizes, int n_in,
                              void* d_out, int out_size, void* d_ws, size_t ws_size,
                              hipStream_t stream) {
  (void)in_sizes; (void)n_in; (void)out_size; (void)ws_size;

  const int*           sentence   = (const int*)d_in[0];
  const int*           tags       = (const int*)d_in[1];
  const unsigned char* mask       = (const unsigned char*)d_in[2];
  const float*         emb        = (const float*)d_in[3];
  const float*         w_ih_f     = (const float*)d_in[4];
  const float*         w_hh_f     = (const float*)d_in[5];
  const float*         b_f        = (const float*)d_in[6];
  const float*         w_ih_b     = (const float*)d_in[7];
  const float*         w_hh_b     = (const float*)d_in[8];
  const float*         b_b        = (const float*)d_in[9];
  const float*         w_out      = (const float*)d_in[10];
  const float*         b_out      = (const float*)d_in[11];
  const float*         start_tr   = (const float*)d_in[12];
  const float*         end_tr     = (const float*)d_in[13];
  const float*         transitions= (const float*)d_in[14];

  char* ws = (char*)d_ws;
  size_t off = 0;
  auto carve = [&](size_t bytes) -> char* {
    char* p = ws + off;
    off = (off + bytes + 255) & ~(size_t)255;
    return p;
  };

  __bf16* x_bf   = (__bf16*)carve(LB * Ec * 2);
  float*  xprojF = (float*) carve(LB * (size_t)G4 * 4);
  float*  xprojB = (float*) carve(LB * (size_t)G4 * 4);
  __bf16* hfall  = (__bf16*)carve(LB * Hc * 2);
  __bf16* hball  = (__bf16*)carve(LB * Hc * 2);
  __bf16* wihF   = (__bf16*)carve((size_t)G4 * Ec * 2);
  __bf16* wihB   = (__bf16*)carve((size_t)G4 * Ec * 2);
  __bf16* whhF   = (__bf16*)carve((size_t)G4 * Hc * 2);
  __bf16* whhB   = (__bf16*)carve((size_t)G4 * Hc * 2);
  __bf16* woutp  = (__bf16*)carve((size_t)16 * HDc * 2);
  __bf16* hFst   = (__bf16*)carve((size_t)Bc * Hc * 2);
  __bf16* hBst   = (__bf16*)carve((size_t)Bc * Hc * 2);
  float*  cF     = (float*) carve((size_t)Bc * Hc * 4);
  float*  cB     = (float*) carve((size_t)Bc * Hc * 4);
  float*  gF     = (float*) carve((size_t)Bc * G4 * 4);
  float*  gB     = (float*) carve((size_t)Bc * G4 * 4);
  float*  feats  = (float*) carve(LB * Tc * 4);

  // --- prep: weight conversion, state init, embedding gather ---
  f32_to_bf16_kernel<<<256, 256, 0, stream>>>(w_ih_f, wihF, G4 * Ec);
  f32_to_bf16_kernel<<<256, 256, 0, stream>>>(w_ih_b, wihB, G4 * Ec);
  f32_to_bf16_kernel<<<256, 256, 0, stream>>>(w_hh_f, whhF, G4 * Hc);
  f32_to_bf16_kernel<<<256, 256, 0, stream>>>(w_hh_b, whhB, G4 * Hc);
  wout_pad_kernel<<<64, 256, 0, stream>>>(w_out, woutp);
  init_state_kernel<<<64, 256, 0, stream>>>(hFst, hBst, cF, cB);
  embed_kernel<<<2048, 256, 0, stream>>>(sentence, emb, x_bf);

  // --- all-timestep input projections (one big WMMA GEMM, both dirs) ---
  xproj_kernel<<<dim3(8, (unsigned)(LB / 32), 2), 256, 0, stream>>>(
      x_bf, wihF, wihB, b_f, b_b, xprojF, xprojB);

  // --- sequential recurrence; fwd & bwd batched per launch ---
  for (int s = 0; s < Lc; ++s) {
    lstm_gates_kernel<<<dim3(G4 / 32, Bc / 32, 2), 32, 0, stream>>>(
        xprojF, xprojB, hFst, hBst, whhF, whhB, gF, gB, s);
    lstm_cell_kernel<<<dim3((Bc * Hc) / 256, 2), 256, 0, stream>>>(
        gF, gB, cF, cB, hFst, hBst, hfall, hball, s);
  }

  // --- output projection + CRF ---
  feats_kernel<<<(unsigned)(LB / 32), 32, 0, stream>>>(hfall, hball, woutp,
                                                       b_out, feats);
  crf_kernel<<<1, Bc, 0, stream>>>(feats, tags, mask, start_tr, end_tr,
                                   transitions, (float*)d_out);
}
